// QuantizedLinear_35716948033552
// MI455X (gfx1250) — matrix operand order repaired and checked by reading, not compiled, not run on hardware
//
#include <hip/hip_runtime.h>
#include <stdint.h>

#define IN_F   8192
#define OUT_F  8192
#define BATCH  32
#define HALF_O 4096   // OUT_F/2 : nibble pairing distance
#define GPR    64     // groups per weight row = IN_F/128

typedef __attribute__((ext_vector_type(16))) _Float16     v16h;
typedef __attribute__((ext_vector_type(8)))  _Float16     v8h;
typedef __attribute__((ext_vector_type(2)))  _Float16     h2;
typedef __attribute__((ext_vector_type(8)))  float        v8f;
typedef __attribute__((ext_vector_type(4)))  float        v4f;
typedef __attribute__((ext_vector_type(4)))  unsigned int v4u;
typedef __attribute__((ext_vector_type(8)))  unsigned int v8u;

// Two int4 values sitting in bits [3:0] (with garbage above) of the two 16-bit
// slots of a dword -> two dequantized f16 weights:
//   h = bitcast<2xf16>((nib & 0x000F000F) | 0x64006400)  == 1024 + q   (1 bitop3)
//   t = h + (-1032)                                      == q - 8      (exact)
//   w = fma(t, s, b),  b = (8 - z) * s                   (packed fused fma)
__device__ __forceinline__ unsigned dq2(unsigned nib2, unsigned m, unsigned g,
                                        h2 s, h2 b, h2 cm) {
    h2 h = __builtin_bit_cast(h2, (nib2 & m) | g);
    h2 w = __builtin_elementwise_fma(h + cm, s, b);
    return __builtin_bit_cast(unsigned, w);
}

struct P2 { v4u a, b; };
// Load A fragment from pre-converted f16 x: halves [kA..kA+7] and [kA+8..kA+15]
__device__ __forceinline__ v16h loadA16(const _Float16* p) {
    P2 s;
    s.a = *(const v4u*)(p);        // 16 B
    s.b = *(const v4u*)(p + 8);    // 16 B
    return __builtin_bit_cast(v16h, s);
}

// Each block: 16 column-pairs (cols c0..c0+15 and c0+4096..), full K.
// Each wave: K slice of 1024 as 8 groups x (4 x K=32 WMMA steps).
template <bool XF16>
__global__ __launch_bounds__(256) void qlin_wmma_kernel(
    const float* __restrict__ x,        // [32, 8192] f32
    const _Float16* __restrict__ xh,    // [32, 8192] f16 (ws), valid iff XF16
    const uint8_t* __restrict__ packed, // 4096*8192 bytes; byte(o%4096,k): hi->row o, lo->row o+4096
    const float* __restrict__ scale,    // [524288]
    const float* __restrict__ zero,     // [524288]
    const float* __restrict__ bias,     // [8192]
    float* __restrict__ out)            // [32, 8192] f32
{
    __shared__ float red[8 * 1024];     // 32 KB: [wave][acc(4)][vgpr(8)][lane(32)]

    const int tid      = threadIdx.x;
    const int wave     = tid >> 5;
    const int lane     = tid & 31;
    const int lane16   = lane & 15;
    const int laneHalf = lane >> 4;     // 0 = lanes 0-15, 1 = lanes 16-31

    const int c0    = blockIdx.x * 16;  // low-half column base
    const int n     = c0 + lane16;      // this lane's packed row / low column
    const int kBase = wave * 1024;      // this wave's K slice

    const uint8_t*  wrow   = packed + (size_t)n * IN_F;
    const float*    xrow0  = x  + (size_t)(lane16)      * IN_F;
    const float*    xrow1  = x  + (size_t)(16 + lane16) * IN_F;
    const _Float16* xhrow0 = xh + (size_t)(lane16)      * IN_F;
    const _Float16* xhrow1 = xh + (size_t)(16 + lane16) * IN_F;

    v8f acc00 = {}, acc01 = {}, acc10 = {}, acc11 = {};  // [mtile][half]

    const _Float16 cmv = (_Float16)(-1032.0f);
    const h2 cm = { cmv, cmv };
    // Keep bitop3 constants in registers so (a&m)|g folds into one op.
    const unsigned msk = 0x000F000Fu;
    const unsigned ebs = 0x64006400u;

    #pragma unroll 1
    for (int grp = 0; grp < 8; ++grp) {
        const int kg0 = kBase + grp * 128;

        // Per-128-group affine params, folded to packed f16 constants.
        const int kg = kg0 >> 7;
        const int g0 = n * GPR + kg;
        const int g1 = (n + HALF_O) * GPR + kg;
        const float sc0 = scale[g0], z0 = zero[g0];
        const float sc1 = scale[g1], z1 = zero[g1];
        h2 s0h, b0h, s1h, b1h;
        s0h[0] = s0h[1] = (_Float16)sc0;
        b0h[0] = b0h[1] = (_Float16)((8.0f - z0) * sc0);
        s1h[0] = s1h[1] = (_Float16)sc1;
        b1h[0] = b1h[1] = (_Float16)((8.0f - z1) * sc1);
        if (kg0 + 512 < IN_F)
            __builtin_prefetch(wrow + kg0 + 512, 0, 1);   // global_prefetch_b8

        #pragma unroll
        for (int j = 0; j < 4; ++j) {
            const int k0 = kg0 + j * 32;

            // ---- B fragments: one b128 = 16 packed bytes per lane, feeds BOTH halves.
            // B 32x16 f16: lanes 0-15 hold K=k0..k0+15, lanes 16-31 hold K=k0+16..k0+31.
            const v4u wq = *(const v4u*)(wrow + k0 + laneHalf * 16);
            v8u b0u, b1u;
            #pragma unroll
            for (int d = 0; d < 4; ++d) {
                const unsigned u  = wq[d];
                const unsigned uh = u >> 4;
                // Spread bytes into 16-bit slots (sel 0xC = constant 0x00);
                // garbage above bit 3 is cleared by the bitop3 in dq2.
                const unsigned p0 = __builtin_amdgcn_perm(uh, uh, 0x0C010C00u);
                const unsigned p1 = __builtin_amdgcn_perm(uh, uh, 0x0C030C02u);
                const unsigned q0 = __builtin_amdgcn_perm(u,  u,  0x0C010C00u);
                const unsigned q1 = __builtin_amdgcn_perm(u,  u,  0x0C030C02u);
                b0u[2 * d]     = dq2(p0, msk, ebs, s0h, b0h, cm);
                b0u[2 * d + 1] = dq2(p1, msk, ebs, s0h, b0h, cm);
                b1u[2 * d]     = dq2(q0, msk, ebs, s1h, b1h, cm);
                b1u[2 * d + 1] = dq2(q1, msk, ebs, s1h, b1h, cm);
            }
            const v16h bf0 = __builtin_bit_cast(v16h, b0u);
            const v16h bf1 = __builtin_bit_cast(v16h, b1u);

            // ---- A fragments.
            // A 16x32 f16: lanes 0-15: K=k0+{0..15}; lanes 16-31: K=k0+{16..31}.
            const int kA = k0 + laneHalf * 16;
            v16h af0, af1;
            if constexpr (XF16) {
                af0 = loadA16(xhrow0 + kA);
                af1 = loadA16(xhrow1 + kA);
            } else {
                const v4f xa0 = *(const v4f*)(xrow0 + kA);
                const v4f xa1 = *(const v4f*)(xrow0 + kA + 4);
                const v4f xa2 = *(const v4f*)(xrow0 + kA + 8);
                const v4f xa3 = *(const v4f*)(xrow0 + kA + 12);
                const v4f xb0 = *(const v4f*)(xrow1 + kA);
                const v4f xb1 = *(const v4f*)(xrow1 + kA + 4);
                const v4f xb2 = *(const v4f*)(xrow1 + kA + 8);
                const v4f xb3 = *(const v4f*)(xrow1 + kA + 12);
                #pragma unroll
                for (int i = 0; i < 4; ++i) {
                    af0[i]      = (_Float16)xa0[i];
                    af0[4 + i]  = (_Float16)xa1[i];
                    af0[8 + i]  = (_Float16)xa2[i];
                    af0[12 + i] = (_Float16)xa3[i];
                    af1[i]      = (_Float16)xb0[i];
                    af1[4 + i]  = (_Float16)xb1[i];
                    af1[8 + i]  = (_Float16)xb2[i];
                    af1[12 + i] = (_Float16)xb3[i];
                }
            }

            // ---- 4 WMMAs: D = A x B + C, f32 accumulate (co-executes with dequant VALU).
            acc00 = __builtin_amdgcn_wmma_f32_16x16x32_f16(false, af0, false, bf0, (short)0, acc00, false, false);
            acc01 = __builtin_amdgcn_wmma_f32_16x16x32_f16(false, af0, false, bf1, (short)0, acc01, false, false);
            acc10 = __builtin_amdgcn_wmma_f32_16x16x32_f16(false, af1, false, bf0, (short)0, acc10, false, false);
            acc11 = __builtin_amdgcn_wmma_f32_16x16x32_f16(false, af1, false, bf1, (short)0, acc11, false, false);
        }
    }

    // ---- Cross-wave K reduction through LDS.
    #pragma unroll
    for (int v = 0; v < 8; ++v) {
        red[wave * 1024 + 0 * 256 + v * 32 + lane] = acc00[v];
        red[wave * 1024 + 1 * 256 + v * 32 + lane] = acc01[v];
        red[wave * 1024 + 2 * 256 + v * 32 + lane] = acc10[v];
        red[wave * 1024 + 3 * 256 + v * 32 + lane] = acc11[v];
    }
    __syncthreads();

    #pragma unroll
    for (int j = 0; j < 4; ++j) {
        const int slot = tid + j * 256;  // acc*256 + vgpr*32 + lane
        float sum = 0.f;
        #pragma unroll
        for (int w = 0; w < 8; ++w) sum += red[w * 1024 + slot];

        const int a     = slot >> 8;         // a = mtile*2 + half
        const int v     = (slot >> 5) & 7;
        const int l     = slot & 31;
        const int mtile = a >> 1;
        const int half  = a & 1;
        // C/D layout: VGPR v, lanes 0-15 -> M=v, N=lane; lanes 16-31 -> M=v+8, N=lane-16
        const int m  = mtile * 16 + v + ((l >> 4) << 3);
        const int nn = c0 + (l & 15) + half * HALF_O;
        out[(size_t)m * OUT_F + nn] = sum + bias[nn];
    }
}

// One-shot x f32 -> f16 conversion into workspace (removes 256x redundant
// per-block conversion work from the hot loop; halves A-side load traffic).
__global__ __launch_bounds__(256) void xcvt_kernel(const float* __restrict__ x,
                                                   _Float16* __restrict__ xh) {
    const int i = (blockIdx.x * 256 + threadIdx.x) * 8;
    const v4f a = *(const v4f*)(x + i);
    const v4f b = *(const v4f*)(x + i + 4);
    v8h h;
    #pragma unroll
    for (int j = 0; j < 4; ++j) {
        h[j]     = (_Float16)a[j];
        h[4 + j] = (_Float16)b[j];
    }
    *(v8h*)(xh + i) = h;
}

extern "C" void kernel_launch(void* const* d_in, const int* in_sizes, int n_in,
                              void* d_out, int out_size, void* d_ws, size_t ws_size,
                              hipStream_t stream) {
    const float*   x      = (const float*)d_in[0];
    const uint8_t* packed = (const uint8_t*)d_in[1];
    const float*   scale  = (const float*)d_in[2];
    const float*   zero   = (const float*)d_in[3];
    const float*   bias   = (const float*)d_in[4];
    float*         out    = (float*)d_out;
    (void)in_sizes; (void)n_in; (void)out_size;

    const size_t xhBytes = (size_t)BATCH * IN_F * sizeof(_Float16);  // 512 KB
    dim3 grid(HALF_O / 16);   // 256 blocks: each owns 16 column-pairs
    dim3 block(256);          // 8 wave32s: K split 8-way, LDS reduce

    if (ws_size >= xhBytes) {
        _Float16* xh = (_Float16*)d_ws;
        xcvt_kernel<<<dim3(BATCH * IN_F / (256 * 8)), block, 0, stream>>>(x, xh);
        qlin_wmma_kernel<true><<<grid, block, 0, stream>>>(x, xh, packed, scale, zero, bias, out);
    } else {
        qlin_wmma_kernel<false><<<grid, block, 0, stream>>>(x, (const _Float16*)x, packed, scale, zero, bias, out);
    }
}
